// MHA_22256520528466
// MI455X (gfx1250) — matrix operand order repaired and checked by reading, not compiled, not run on hardware
//
#include <hip/hip_runtime.h>

// ---------------------------------------------------------------------------
// MHA for MI455X (gfx1250): bf16 WMMA end-to-end, f32 accumulate.
// B=2, D=1024, T=2048, H=16, Dh=64.
//
// ~60 GFLOP vs ~46MB HBM => matrix-pipe bound; all tensors are L2-resident.
// v4: same as v3 but use __builtin_amdgcn_exp2f (native v_exp_f32) instead of
//     __exp2f, which this toolchain resolves to the glibc host function.
// Design: (1/sqrt(Dh))*log2(e) folded into the Q fragment at preload so the
// attention inner loop has no scaling multiplies; branch-free epilogues;
// double-buffered LDS with a register-staged pipeline (one barrier per tile,
// next-tile global loads overlap WMMAs); K/V tiles staged in LDS shared by
// all 8 waves; packed v2bf (b32) LDS staging stores.
// Fragment layouts (k order used by this code, same for A and B):
//   A 16x32 bf16: lane m=L&15; elems 0-15 <-> k=16h+e
//   B 32x16 bf16: lane n=L&15; elems 0-15 <-> k=16h+e  (N-major "Bt" storage)
// Q,K stored (T,Dh) row-major; V stored (Dh,T); attention computes S^T / O^T
// so softmax state is per-lane and P^T needs only xor-16 shuffles.
// ---------------------------------------------------------------------------

typedef __attribute__((ext_vector_type(16))) __bf16 v16bf;
typedef __attribute__((ext_vector_type(8)))  __bf16 v8bf;
typedef __attribute__((ext_vector_type(2)))  __bf16 v2bf;
typedef __attribute__((ext_vector_type(8)))  float  v8f;
typedef __attribute__((ext_vector_type(4)))  float  v4f;

#define WMMA_BF16(A, Bm, C) \
  __builtin_amdgcn_wmma_f32_16x16x32_bf16(false, (A), false, (Bm), (short)0, (C), false, false)

#define EXP2F(x) __builtin_amdgcn_exp2f(x)   // native v_exp_f32

constexpr int Bc = 2, Dc = 1024, Tc = 2048, Hc = 16, DHc = 64;
constexpr int BM = 128;   // t-tile rows per block (kernel 1)
constexpr int BK = 32;    // k-step (= WMMA K for bf16)

static __device__ __forceinline__ v16bf cat8(v8bf lo, v8bf hi) {
  return __builtin_shufflevector(lo, hi, 0,1,2,3,4,5,6,7,8,9,10,11,12,13,14,15);
}

// A-fragment (16x32 bf16) from a row-major MxK row pointer: two 16B loads.
static __device__ __forceinline__ v16bf load_afrag(const __bf16* row_k0) {
  v8bf lo = *(const v8bf*)(row_k0);        // k = 16h + e     (e=0..7)
  v8bf hi = *(const v8bf*)(row_k0 + 8);    // k = 16h + e     (e=8..15)
  return cat8(lo, hi);
}

// ===========================================================================
// Kernel 1: fused QKV projection.
//   proj[n,b,h,t,dh] = sum_d x[b,d,t] * w[n,h,d,dh]
// One block = one (n,b,h) and a 128-row t-tile; 8 waves x 16 rows each.
// Double-buffered LDS, one barrier per 32-wide k-step.
// ===========================================================================
__global__ __launch_bounds__(256)
void qkv_proj_kernel(const float* __restrict__ x, const float* __restrict__ w,
                     __bf16* __restrict__ qp, __bf16* __restrict__ kp,
                     __bf16* __restrict__ vp)
{
  __shared__ __bf16 Al[2][BM * BK];   // [t_local][d_local]   (M x K)  8KB each
  __shared__ __bf16 Wl[2][DHc * BK];  // [dh][d_local]        (N x K)  4KB each

  const int tid  = threadIdx.x;
  const int tb   = blockIdx.x * BM;
  const int nbh  = blockIdx.y;                 // 0..95
  const int nmat = nbh / (Bc * Hc);            // 0=q 1=k 2=v
  const int b    = (nbh / Hc) % Bc;
  const int hh   = nbh % Hc;

  const int lane = tid & 31, wv = tid >> 5;
  const int h = lane >> 4, nl = lane & 15;

  const float* xb = x + (size_t)b * Dc * Tc + tb;
  const float* wb = w + (size_t)(nmat * Hc + hh) * Dc * DHc;

  // staging decomposition (transpose happens at LDS-write time, packed b32)
  const int tq = tid & 31;          // x: t-quad (t = tq*4 + i)
  const int ap = (tid >> 5) * 2;    // x: first d-pair index (0,2,..,14)
  const int dq = tid & 15;          // W: dh-quad
  const int cp = tid >> 4;          // W: d-pair (0..15)

  v4f xr[4];   // x rows d = 2(ap+j)+k for j,k in {0,1}
  v4f wr[2];   // w rows d = 2cp, 2cp+1

  auto stage_load = [&](int d0) {
#pragma unroll
    for (int j = 0; j < 2; ++j) {
      const int d = (ap + j) * 2;
      xr[j * 2 + 0] = *(const v4f*)(xb + (size_t)(d0 + d + 0) * Tc + tq * 4);
      xr[j * 2 + 1] = *(const v4f*)(xb + (size_t)(d0 + d + 1) * Tc + tq * 4);
    }
    wr[0] = *(const v4f*)(wb + (size_t)(d0 + cp * 2 + 0) * DHc + dq * 4);
    wr[1] = *(const v4f*)(wb + (size_t)(d0 + cp * 2 + 1) * DHc + dq * 4);
  };
  auto stage_store = [&](int p) {
    __bf16* A = Al[p];
    __bf16* W = Wl[p];
#pragma unroll
    for (int j = 0; j < 2; ++j) {
      const int d = (ap + j) * 2;
#pragma unroll
      for (int i = 0; i < 4; ++i) {
        v2bf pk = {(__bf16)xr[j * 2 + 0][i], (__bf16)xr[j * 2 + 1][i]};
        *(v2bf*)&A[(tq * 4 + i) * BK + d] = pk;        // Al[t][d], d-pair
      }
    }
#pragma unroll
    for (int i = 0; i < 4; ++i) {
      v2bf pk = {(__bf16)wr[0][i], (__bf16)wr[1][i]};
      *(v2bf*)&W[(dq * 4 + i) * BK + cp * 2] = pk;     // Wl[dh][d], d-pair
    }
  };

  v8f ct[4] = {v8f{}, v8f{}, v8f{}, v8f{}};

  stage_load(0);
  for (int d0 = 0; d0 < Dc; d0 += BK) {
    const int p = (d0 >> 5) & 1;
    stage_store(p);
    __syncthreads();
    if (d0 + BK < Dc) stage_load(d0 + BK);   // overlaps the WMMAs below

    const v16bf a = load_afrag(&Al[p][(wv * 16 + nl) * BK + 16 * h]);
#pragma unroll
    for (int nt = 0; nt < 4; ++nt) {
      const v16bf bm = *(const v16bf*)&Wl[p][(nt * 16 + nl) * BK + 16 * h];
      ct[nt] = WMMA_BF16(a, bm, ct[nt]);
    }
  }

  // ---- branch-free epilogue: pick (dst, base, strides) once ----------------
  const size_t bh = (size_t)b * Hc + hh;
  __bf16* dst;
  size_t base;
  int ms, ns;
  if (nmat == 2) { dst = vp; base = bh * DHc * Tc; ms = 1;   ns = Tc; } // (Dh,T)
  else {
    dst = (nmat == 0) ? qp : kp;
    base = bh * Tc * DHc; ms = DHc; ns = 1;                             // (T,Dh)
  }
#pragma unroll
  for (int nt = 0; nt < 4; ++nt) {
#pragma unroll
    for (int r = 0; r < 8; ++r) {
      const int m = tb + wv * 16 + r + 8 * h;   // t index
      const int n = nt * 16 + nl;               // dh index
      dst[base + (size_t)m * ms + (size_t)n * ns] = (__bf16)ct[nt][r];
    }
  }
}

// ===========================================================================
// Kernel 2: flash attention per (b,h), transposed formulation.
//   S^T[key][query] = K row (A) x Q (B);  O^T[dh][query] = V^T row (A) x P^T (B)
// Wave owns 16 queries (lane = query). 64 key-tile iterations of 32 keys.
// K/V tiles double-buffered in LDS, shared by all 8 waves; one barrier/tile.
// Softmax runs in base-2: (1/sqrt(Dh))*log2(e) is pre-folded into Q.
// ===========================================================================
__global__ __launch_bounds__(256)
void attn_kernel(const __bf16* __restrict__ qp, const __bf16* __restrict__ kp,
                 const __bf16* __restrict__ vp, float* __restrict__ out)
{
  __shared__ __bf16 Kl[2][32 * DHc];   // [key_local][dh]   4KB each
  __shared__ __bf16 Vl[2][DHc * 32];   // [dh][key_local]   4KB each

  const int tid  = threadIdx.x;
  const int lane = tid & 31, wv = tid >> 5;
  const int h = lane >> 4, nl = lane & 15;
  const int qb0 = blockIdx.x * 128 + wv * 16;   // query base for this wave
  const int bh  = blockIdx.y;                   // b*H + h

  const __bf16* qh = qp + (size_t)bh * Tc * DHc;
  const __bf16* kh = kp + (size_t)bh * Tc * DHc;
  const __bf16* vh = vp + (size_t)bh * DHc * Tc;

  // K stage: 32 rows x 128B ; V stage: 64 rows x 64B ; 16B per thread each
  const int krow = tid >> 3, kseg = tid & 7;
  const int vrow = tid >> 2, vseg = tid & 3;

  v8bf kst, vst;
  auto stage_load = [&](int kb) {
    kst = *(const v8bf*)(kh + (size_t)(kb + krow) * DHc + kseg * 8);
    vst = *(const v8bf*)(vh + (size_t)vrow * Tc + kb + vseg * 8);
  };
  auto stage_store = [&](int p) {
    *(v8bf*)&Kl[p][krow * DHc + kseg * 8] = kst;
    *(v8bf*)&Vl[p][vrow * 32 + vseg * 8] = vst;
  };

  // Q as B-fragment, preloaded once; fold softmax scale into Q so the inner
  // loop needs no scaling multiplies and can use native exp2.
  constexpr float kQScale = 0.125f * 1.44269504088896340736f;  // 1/sqrt(64)*log2(e)
  v16bf qreg[2];
#pragma unroll
  for (int kk = 0; kk < 2; ++kk) {
    qreg[kk] = *(const v16bf*)(qh + (size_t)(qb0 + nl) * DHc + kk * 32 + 16 * h);
#pragma unroll
    for (int e = 0; e < 16; ++e)
      qreg[kk][e] = (__bf16)((float)qreg[kk][e] * kQScale);
  }

  float mi = -__builtin_inff();
  float li = 0.0f;
  v8f ot[4] = {v8f{}, v8f{}, v8f{}, v8f{}};

  stage_load(0);
  for (int it = 0; it < Tc / 32; ++it) {
    const int p = it & 1;
    stage_store(p);
    __syncthreads();
    if (it + 1 < Tc / 32) stage_load((it + 1) * 32);   // overlaps compute

    // ---- S^T tiles (already in log2 domain): keys 0..15 / 16..31 -----------
    v8f st0{}, st1{};
#pragma unroll
    for (int kk = 0; kk < 2; ++kk) {
      const v16bf a0 = load_afrag(&Kl[p][(nl)      * DHc + kk * 32 + 16 * h]);
      const v16bf a1 = load_afrag(&Kl[p][(16 + nl) * DHc + kk * 32 + 16 * h]);
      st0 = WMMA_BF16(a0, qreg[kk], st0);
      st1 = WMMA_BF16(a1, qreg[kk], st1);
    }

    // ---- online softmax, base-2 (per-lane state: lane = query) -------------
    float tmax = -__builtin_inff();
#pragma unroll
    for (int r = 0; r < 8; ++r)
      tmax = fmaxf(tmax, fmaxf(st0[r], st1[r]));
    tmax = fmaxf(tmax, __shfl_xor(tmax, 16, 32));   // join the two key halves
    const float mnew  = fmaxf(mi, tmax);
    const float alpha = EXP2F(mi - mnew);

    float p0[8], p1[8];
    float rsum = 0.0f;
#pragma unroll
    for (int r = 0; r < 8; ++r) {
      p0[r] = EXP2F(st0[r] - mnew);
      p1[r] = EXP2F(st1[r] - mnew);
      rsum += p0[r] + p1[r];
    }
    rsum += __shfl_xor(rsum, 16, 32);
    li = li * alpha + rsum;
    mi = mnew;

    // ---- build P^T B-fragment: lane n = query, elem e <-> key = 16h+e ------
    // D-layout holds key = r + 8h, so the missing 8 keys live in lane^16.
    v16bf pt{};
#pragma unroll
    for (int e = 0; e < 8; ++e) {
      const float o0 = __shfl_xor(p0[e], 16, 32);
      const float o1 = __shfl_xor(p1[e], 16, 32);
      pt[e]     = (__bf16)(h ? o1 : p0[e]);   // keys 16h + 0..7
      pt[e + 8] = (__bf16)(h ? p1[e] : o0);   // keys 16h + 8..15
    }

    // ---- O^T += V^T x P^T (rescale by alpha first) -------------------------
#pragma unroll
    for (int t4 = 0; t4 < 4; ++t4) {
#pragma unroll
      for (int r = 0; r < 8; ++r) ot[t4][r] *= alpha;
      const v16bf va = load_afrag(&Vl[p][(t4 * 16 + nl) * 32 + 16 * h]);
      ot[t4] = WMMA_BF16(va, pt, ot[t4]);
    }
  }

  // ---- normalize + store; O^T D-layout is exactly out (B, H*Dh, T) ---------
  const float inv = 1.0f / li;
  const int b = bh / Hc, hh = bh % Hc;
  float* ob = out + ((size_t)b * Dc + hh * DHc) * Tc;
#pragma unroll
  for (int t4 = 0; t4 < 4; ++t4) {
#pragma unroll
    for (int r = 0; r < 8; ++r) {
      const int dh = t4 * 16 + r + 8 * h;
      ob[(size_t)dh * Tc + qb0 + nl] = ot[t4][r] * inv;
    }
  }
}

// ===========================================================================
extern "C" void kernel_launch(void* const* d_in, const int* in_sizes, int n_in,
                              void* d_out, int out_size, void* d_ws, size_t ws_size,
                              hipStream_t stream) {
  const float* x   = (const float*)d_in[0];   // (B, D, T) f32
  const float* qkv = (const float*)d_in[1];   // (3, H, D, Dh) f32

  const size_t per = (size_t)Bc * Hc * Tc * DHc;      // 4,194,304 elems
  if (ws_size < 3 * per * sizeof(__bf16)) return;     // 25.2 MB scratch needed

  __bf16* qp = (__bf16*)d_ws;
  __bf16* kp = qp + per;
  __bf16* vp = kp + per;

  dim3 g1(Tc / BM, 3 * Bc * Hc);   // 16 x 96
  qkv_proj_kernel<<<g1, 256, 0, stream>>>(x, qkv, qp, kp, vp);

  dim3 g2(Tc / 128, Bc * Hc);      // 16 x 32
  attn_kernel<<<g2, 256, 0, stream>>>(qp, kp, vp, (float*)d_out);
}
